// GatedDeltaNet_69355131896515
// MI455X (gfx1250) — matrix operand order repaired and checked by reading, not compiled, not run on hardware
//
#include <hip/hip_runtime.h>
#include <math.h>

#define B_ 2
#define T_ 2048
#define D_ 1024
#define H_ 16
#define HD_ 64
#define KCONV 4
#define EPS_ 1e-6f

typedef __bf16 v16bf __attribute__((ext_vector_type(16)));
typedef float  v8f   __attribute__((ext_vector_type(8)));
typedef float  f32x4 __attribute__((ext_vector_type(4)));

union FragAB { f32x4 f4[2]; v16bf v; };

// ---------------- fp32 -> bf16 conversion ----------------
__global__ void f32_to_bf16_kernel(const float* __restrict__ in,
                                   __bf16* __restrict__ out, int n) {
  int i = blockIdx.x * blockDim.x + threadIdx.x;
  int stride = gridDim.x * blockDim.x;
  for (; i < n; i += stride) out[i] = (__bf16)in[i];
}

// ---------------- WMMA GEMM: C[M,N] = A[M,K] * B[N,K]^T ----------------
// A, B bf16 row-major; C fp32 row-major. M%64==0, N%256==0, K%32==0.
// Block: 256 threads = 8 waves in a 2(M) x 4(N) grid; 32x64 tile per wave.
__global__ void __launch_bounds__(256)
wmma_gemm_bf16(const __bf16* __restrict__ A, const __bf16* __restrict__ Bw,
               float* __restrict__ C, int M, int N, int Kd) {
  const int lane  = threadIdx.x & 31;
  const int wave  = threadIdx.x >> 5;
  const int waveM = wave & 1;
  const int waveN = wave >> 1;
  const int half  = lane >> 4;   // 0/1
  const int l16   = lane & 15;
  const int m0 = blockIdx.x * 64 + waveM * 32;
  const int n0 = blockIdx.y * 256 + waveN * 64;

  v8f acc[2][4];
#pragma unroll
  for (int i = 0; i < 2; i++)
#pragma unroll
    for (int j = 0; j < 4; j++)
#pragma unroll
      for (int e = 0; e < 8; e++) acc[i][j][e] = 0.0f;

  for (int kb = 0; kb < Kd; kb += 32) {
    FragAB a[2], b[4];
#pragma unroll
    for (int i = 0; i < 2; i++) {
      const __bf16* ap = A + (long)(m0 + i * 16 + l16) * Kd + kb;
      a[i].f4[0] = *(const f32x4*)(ap + 8 * half);         // K = 8*half .. +7
      a[i].f4[1] = *(const f32x4*)(ap + 16 + 8 * half);    // K = 16+8*half .. +7
    }
#pragma unroll
    for (int j = 0; j < 4; j++) {
      const __bf16* bp = Bw + (long)(n0 + j * 16 + l16) * Kd + kb;
      b[j].f4[0] = *(const f32x4*)(bp + 8 * half);         // K = 8*half .. +7
      b[j].f4[1] = *(const f32x4*)(bp + 16 + 8 * half);    // K = 16+8*half .. +7
    }
#pragma unroll
    for (int i = 0; i < 2; i++)
#pragma unroll
      for (int j = 0; j < 4; j++)
        acc[i][j] = __builtin_amdgcn_wmma_f32_16x16x32_bf16(
            false, a[i].v, false, b[j].v, (short)0, acc[i][j], false, false);
  }

#pragma unroll
  for (int i = 0; i < 2; i++)
#pragma unroll
    for (int j = 0; j < 4; j++)
#pragma unroll
      for (int v = 0; v < 8; v++) {
        int row = m0 + i * 16 + v + 8 * half;
        int col = n0 + j * 16 + l16;
        C[(long)row * N + col] = acc[i][j][v];
      }
}

// ---------------- small projections: beta = sigmoid(h@Wb^T), g gate ----------------
__global__ void proj_ab_kernel(const float* __restrict__ h, const float* __restrict__ Wa,
                               const float* __restrict__ Wb, const float* __restrict__ A_log,
                               const float* __restrict__ dt_bias,
                               float* __restrict__ g, float* __restrict__ beta) {
  int idx = blockIdx.x * blockDim.x + threadIdx.x;
  if (idx >= B_ * T_ * H_) return;
  int hh = idx % H_;
  int bt = idx / H_;
  const float* hr = h + (long)bt * D_;
  const float* wa = Wa + hh * D_;
  const float* wb = Wb + hh * D_;
  float sa = 0.f, sb = 0.f;
  for (int d = 0; d < D_; d += 4) {
    f32x4 x  = *(const f32x4*)(hr + d);
    f32x4 a4 = *(const f32x4*)(wa + d);
    f32x4 b4 = *(const f32x4*)(wb + d);
#pragma unroll
    for (int e = 0; e < 4; e++) { sa = fmaf(x[e], a4[e], sa); sb = fmaf(x[e], b4[e], sb); }
  }
  beta[idx] = 1.f / (1.f + __expf(-sb));
  float ap = sa + dt_bias[hh];
  float sp = (ap > 20.f) ? ap : log1pf(__expf(ap));
  g[idx] = -__expf(A_log[hh]) * sp;
}

// ---------------- 64-wide block reduction ----------------
__device__ __forceinline__ float block64_sum(float v, float* s, int lane) {
  s[lane] = v;
  __syncthreads();
#pragma unroll
  for (int off = 32; off > 0; off >>= 1) {
    if (lane < off) s[lane] += s[lane + off];
    __syncthreads();
  }
  float r = s[0];
  __syncthreads();
  return r;
}

// ---------------- causal depthwise conv (K=4) + SiLU + rmsnorm + l2norm ----------------
__global__ void __launch_bounds__(64)
conv_norm_qkv_kernel(const float* __restrict__ qp, const float* __restrict__ kp,
                     const float* __restrict__ vp, const float* __restrict__ cwq,
                     const float* __restrict__ cwk, const float* __restrict__ cwv,
                     const float* __restrict__ qnw, const float* __restrict__ knw,
                     float* __restrict__ qn, float* __restrict__ kn, float* __restrict__ vc) {
  __shared__ float sred[64];
  int bth = blockIdx.x;
  int h = bth % H_;
  int bt = bth / H_;
  int t = bt % T_;
  int lane = threadIdx.x;
  int c = h * HD_ + lane;
  float aq = 0.f, ak = 0.f, av = 0.f;
#pragma unroll
  for (int j = 0; j < KCONV; j++) {
    if (t - (KCONV - 1) + j >= 0) {
      long idx = (long)(bt - (KCONV - 1) + j) * D_ + c;
      aq = fmaf(qp[idx], cwq[c * KCONV + j], aq);
      ak = fmaf(kp[idx], cwk[c * KCONV + j], ak);
      av = fmaf(vp[idx], cwv[c * KCONV + j], av);
    }
  }
  float yq = aq / (1.f + __expf(-aq));
  float yk = ak / (1.f + __expf(-ak));
  float yv = av / (1.f + __expf(-av));
  // q: rmsnorm then l2norm
  float msq = block64_sum(yq * yq, sred, lane) * (1.f / HD_);
  float xq = yq * rsqrtf(msq + EPS_) * qnw[lane];
  float s2q = block64_sum(xq * xq, sred, lane);
  xq = xq * rsqrtf(s2q + EPS_);
  // k: rmsnorm then l2norm
  float msk = block64_sum(yk * yk, sred, lane) * (1.f / HD_);
  float xk = yk * rsqrtf(msk + EPS_) * knw[lane];
  float s2k = block64_sum(xk * xk, sred, lane);
  xk = xk * rsqrtf(s2k + EPS_);
  long o = (long)bt * D_ + c;
  qn[o] = xq; kn[o] = xk; vc[o] = yv;
}

// ---------------- sequential gated delta-rule scan ----------------
// One block per (b,h); 256 threads; thread owns S[r][c] for c = tid%64, r in [rq*16, rq*16+16)
__global__ void __launch_bounds__(256)
scan_kernel(const float* __restrict__ qn, const float* __restrict__ kn,
            const float* __restrict__ vc, const float* __restrict__ g,
            const float* __restrict__ beta, float* __restrict__ o) {
  int bh = blockIdx.x;
  int b = bh / H_;
  int h = bh % H_;
  int tid = threadIdx.x;
  int c = tid & (HD_ - 1);
  int rq = tid >> 6;  // 0..3
  float S[16];
#pragma unroll
  for (int i = 0; i < 16; i++) S[i] = 0.f;
  __shared__ float kq[HD_], qq[HD_], vv[HD_];
  __shared__ float red[4][HD_];
  const float scale = 0.125f;  // 64^-0.5
  for (int t = 0; t < T_; t++) {
    long base = ((long)(b * T_ + t)) * D_ + h * HD_;
    if (tid < HD_) {
      kq[tid] = kn[base + tid];
      qq[tid] = qn[base + tid];
      vv[tid] = vc[base + tid];
    }
    __syncthreads();  // S1
    float gt = g[(b * T_ + t) * H_ + h];
    float bt = beta[(b * T_ + t) * H_ + h];
    float dec = __expf(gt);
    float p = 0.f;
#pragma unroll
    for (int i = 0; i < 16; i++) {
      S[i] *= dec;
      p = fmaf(kq[rq * 16 + i], S[i], p);
    }
    red[rq][c] = p;
    __syncthreads();  // S2
    float kv = red[0][c] + red[1][c] + red[2][c] + red[3][c];
    float dl = (vv[c] - kv) * bt;
    __syncthreads();  // S3 (all done reading red/vv before rewrite)
    float po = 0.f;
#pragma unroll
    for (int i = 0; i < 16; i++) {
      S[i] = fmaf(kq[rq * 16 + i], dl, S[i]);
      po = fmaf(qq[rq * 16 + i] * scale, S[i], po);
    }
    red[rq][c] = po;
    __syncthreads();  // S4
    if (rq == 0) o[base + c] = red[0][c] + red[1][c] + red[2][c] + red[3][c];
  }
}

// ---------------- output rmsnorm * swish(gate), write bf16 for final GEMM ----------------
__global__ void __launch_bounds__(64)
gate_out_kernel(const float* __restrict__ o_scan, const float* __restrict__ g_pre,
                const float* __restrict__ onw, __bf16* __restrict__ og) {
  __shared__ float sred[64];
  int bth = blockIdx.x;
  int h = bth % H_;
  int bt = bth / H_;
  int lane = threadIdx.x;
  int c = h * HD_ + lane;
  long idx = (long)bt * D_ + c;
  float o = o_scan[idx];
  float ms = block64_sum(o * o, sred, lane) * (1.f / HD_);
  float on = o * rsqrtf(ms + EPS_) * onw[lane];
  float gv = g_pre[idx];
  float gate = gv / (1.f + __expf(-gv));
  og[idx] = (__bf16)(on * gate);
}

// ---------------- host launch ----------------
extern "C" void kernel_launch(void* const* d_in, const int* in_sizes, int n_in,
                              void* d_out, int out_size, void* d_ws, size_t ws_size,
                              hipStream_t stream) {
  const float* h       = (const float*)d_in[0];
  const float* Wq      = (const float*)d_in[1];
  const float* Wk      = (const float*)d_in[2];
  const float* Wv      = (const float*)d_in[3];
  const float* Wa      = (const float*)d_in[4];
  const float* Wb      = (const float*)d_in[5];
  const float* Wg      = (const float*)d_in[6];
  const float* Wo      = (const float*)d_in[7];
  const float* q_norm  = (const float*)d_in[8];
  const float* k_norm  = (const float*)d_in[9];
  const float* o_norm  = (const float*)d_in[10];
  const float* A_log   = (const float*)d_in[11];
  const float* dt_bias = (const float*)d_in[12];
  const float* conv_q  = (const float*)d_in[13];
  const float* conv_k  = (const float*)d_in[14];
  const float* conv_v  = (const float*)d_in[15];

  const size_t MT = (size_t)B_ * T_;  // 4096 rows
  char* ws = (char*)d_ws;
  size_t off = 0;
  auto alloc = [&](size_t bytes) {
    size_t o = off;
    off += (bytes + 255) & ~(size_t)255;
    return o;
  };

  __bf16* h_bf   = (__bf16*)(ws + alloc(MT * D_ * 2));
  __bf16* Wq_bf  = (__bf16*)(ws + alloc((size_t)D_ * D_ * 2));
  __bf16* Wk_bf  = (__bf16*)(ws + alloc((size_t)D_ * D_ * 2));
  __bf16* Wv_bf  = (__bf16*)(ws + alloc((size_t)D_ * D_ * 2));
  __bf16* Wg_bf  = (__bf16*)(ws + alloc((size_t)D_ * D_ * 2));
  __bf16* Wo_bf  = (__bf16*)(ws + alloc((size_t)D_ * D_ * 2));
  float* q_pre   = (float*)(ws + alloc(MT * D_ * 4));
  float* k_pre   = (float*)(ws + alloc(MT * D_ * 4));
  float* v_pre   = (float*)(ws + alloc(MT * D_ * 4));
  float* g_pre   = (float*)(ws + alloc(MT * D_ * 4));
  float* qn      = (float*)(ws + alloc(MT * D_ * 4));
  float* kn      = (float*)(ws + alloc(MT * D_ * 4));
  float* vc      = (float*)(ws + alloc(MT * D_ * 4));
  float* gvals   = (float*)(ws + alloc(MT * H_ * 4));
  float* betav   = (float*)(ws + alloc(MT * H_ * 4));
  float* o_scan  = (float*)(ws + alloc(MT * D_ * 4));
  __bf16* og_bf  = (__bf16*)(ws + alloc(MT * D_ * 2));
  (void)ws_size; (void)in_sizes; (void)n_in; (void)out_size;

  // 1) bf16 conversions
  f32_to_bf16_kernel<<<512, 256, 0, stream>>>(h, h_bf, (int)(MT * D_));
  f32_to_bf16_kernel<<<256, 256, 0, stream>>>(Wq, Wq_bf, D_ * D_);
  f32_to_bf16_kernel<<<256, 256, 0, stream>>>(Wk, Wk_bf, D_ * D_);
  f32_to_bf16_kernel<<<256, 256, 0, stream>>>(Wv, Wv_bf, D_ * D_);
  f32_to_bf16_kernel<<<256, 256, 0, stream>>>(Wg, Wg_bf, D_ * D_);
  f32_to_bf16_kernel<<<256, 256, 0, stream>>>(Wo, Wo_bf, D_ * D_);

  // 2) WMMA projections: q/k/v/g = h @ W^T
  dim3 gg((unsigned)(MT / 64), (unsigned)(D_ / 256));
  wmma_gemm_bf16<<<gg, 256, 0, stream>>>(h_bf, Wq_bf, q_pre, (int)MT, D_, D_);
  wmma_gemm_bf16<<<gg, 256, 0, stream>>>(h_bf, Wk_bf, k_pre, (int)MT, D_, D_);
  wmma_gemm_bf16<<<gg, 256, 0, stream>>>(h_bf, Wv_bf, v_pre, (int)MT, D_, D_);
  wmma_gemm_bf16<<<gg, 256, 0, stream>>>(h_bf, Wg_bf, g_pre, (int)MT, D_, D_);

  // 3) beta / g gate scalars
  proj_ab_kernel<<<(B_ * T_ * H_ + 255) / 256, 256, 0, stream>>>(h, Wa, Wb, A_log,
                                                                 dt_bias, gvals, betav);

  // 4) conv + SiLU + norms
  conv_norm_qkv_kernel<<<B_ * T_ * H_, 64, 0, stream>>>(q_pre, k_pre, v_pre, conv_q,
                                                        conv_k, conv_v, q_norm, k_norm,
                                                        qn, kn, vc);

  // 5) sequential delta-rule scan
  scan_kernel<<<B_ * H_, 256, 0, stream>>>(qn, kn, vc, gvals, betav, o_scan);

  // 6) output norm * swish gate -> bf16
  gate_out_kernel<<<B_ * T_ * H_, 64, 0, stream>>>(o_scan, g_pre, o_norm, og_bf);

  // 7) final WMMA GEMM: out = og @ Wo^T (fp32 out)
  wmma_gemm_bf16<<<gg, 256, 0, stream>>>(og_bf, Wo_bf, (float*)d_out, (int)MT, D_, D_);
}
